// MoE_7773890806160
// MI455X (gfx1250) — compile-verified, matrix operand order checked by reading, not run on hardware
//
#include <hip/hip_runtime.h>
#include <hip/hip_bf16.h>

// ---------------------------------------------------------------------------
// MoE (top-2 of 8 experts), N=8192 D=1024 H=2048 O=1024.
// Routed execution: 275 GFLOP (vs 1.1 TFLOP dense). bf16 WMMA, f32 accum.
// GEMM tiling: block = 32 (M) x 256 (N), 8 waves; each wave holds
// 2 A frags x 2 B frags -> 4 v_wmma per K-step (2x data reuse vs 16-row tile).
// A tiles staged to LDS via GLOBAL_LOAD_ASYNC_TO_LDS_B128 (ASYNCcnt path).
// Workspace requirement ~290 MB.
// ---------------------------------------------------------------------------

#define MOE_N 8192
#define MOE_D 1024
#define MOE_H 2048
#define MOE_O 1024
#define MOE_E 8
#define GATE_H 64

typedef __attribute__((ext_vector_type(16))) __bf16 v16bf;
typedef __attribute__((ext_vector_type(8)))  float  v8f;

union FragB16 {
    v16bf v;
    uint4 q[2];
};

#if defined(__has_builtin)
#if __has_builtin(__builtin_amdgcn_global_load_async_to_lds_b128)
#define HAVE_ASYNC_LDS 1
#else
#define HAVE_ASYNC_LDS 0
#endif
#else
#define HAVE_ASYNC_LDS 0
#endif

#if HAVE_ASYNC_LDS
// builtin signature (from hipcc diagnostic): <4 x i32> AS1* src, AS3* dst,
// imm offset, imm cpol
typedef int av4i __attribute__((vector_size(16)));
typedef __attribute__((address_space(1))) av4i* as1_v4i_p;
typedef __attribute__((address_space(3))) av4i* as3_v4i_p;
#endif

__device__ __forceinline__ void async_wait0() {
#if defined(__has_builtin) && __has_builtin(__builtin_amdgcn_s_wait_asynccnt)
    __builtin_amdgcn_s_wait_asynccnt(0);
#else
    asm volatile("s_wait_asynccnt 0" ::: "memory");
#endif
}

// ---------------------------------------------------------------------------
// fp32 -> bf16 elementwise convert
// ---------------------------------------------------------------------------
__global__ void convert_bf16_kernel(const float* __restrict__ src,
                                    __bf16* __restrict__ dst, size_t n) {
    size_t i = (size_t)blockIdx.x * blockDim.x + threadIdx.x;
    if (i < n) dst[i] = (__bf16)src[i];
}

// dst[e][c][r] = (bf16) src[e][r][c]   (per-expert transpose + convert)
__global__ void transpose_bf16_kernel(const float* __restrict__ src,
                                      __bf16* __restrict__ dst,
                                      int E, int R, int C) {
    size_t i = (size_t)blockIdx.x * blockDim.x + threadIdx.x;
    size_t tot = (size_t)E * R * C;
    if (i >= tot) return;
    size_t rc = (size_t)R * C;
    int e = (int)(i / rc);
    size_t rem = i % rc;
    int c = (int)(rem / R);
    int r = (int)(rem % R);
    dst[i] = (__bf16)src[((size_t)e * R + r) * C + c];
}

// ---------------------------------------------------------------------------
// Gating: one 64-thread block per token.
// ---------------------------------------------------------------------------
__global__ void gate_kernel(const float* __restrict__ x,
                            const float* __restrict__ Wg1,
                            const float* __restrict__ bg1,
                            const float* __restrict__ Wg2,
                            const float* __restrict__ bg2,
                            int* __restrict__ counts,
                            int* __restrict__ topE,
                            float* __restrict__ topW) {
    __shared__ float xs[MOE_D];
    __shared__ float g1s[GATE_H];
    __shared__ float lg[MOE_E];
    int n = blockIdx.x;
    int t = threadIdx.x;  // 0..63
    for (int i = t; i < MOE_D; i += GATE_H)
        xs[i] = x[(size_t)n * MOE_D + i];
    __syncthreads();
    float acc = bg1[t];
    for (int d = 0; d < MOE_D; ++d)
        acc += xs[d] * Wg1[d * GATE_H + t];
    g1s[t] = fmaxf(acc, 0.0f);
    __syncthreads();
    if (t < MOE_E) {
        float a = bg2[t];
        for (int j = 0; j < GATE_H; ++j)
            a += g1s[j] * Wg2[j * MOE_E + t];
        lg[t] = a;
    }
    __syncthreads();
    if (t == 0) {
        float mx = lg[0];
        for (int e = 1; e < MOE_E; ++e) mx = fmaxf(mx, lg[e]);
        float p[MOE_E], s = 0.0f;
        for (int e = 0; e < MOE_E; ++e) { p[e] = __expf(lg[e] - mx); s += p[e]; }
        float inv = 1.0f / s;
        for (int e = 0; e < MOE_E; ++e) p[e] *= inv;
        int i0 = 0;
        for (int e = 1; e < MOE_E; ++e) if (p[e] > p[i0]) i0 = e;
        int i1 = (i0 == 0) ? 1 : 0;
        for (int e = 0; e < MOE_E; ++e)
            if (e != i0 && p[e] > p[i1]) i1 = e;
        topE[n * 2 + 0] = i0; topW[n * 2 + 0] = p[i0];
        topE[n * 2 + 1] = i1; topW[n * 2 + 1] = p[i1];
        atomicAdd(&counts[i0], 1);
        atomicAdd(&counts[i1], 1);
    }
}

// exclusive scan over E=8 counts (single thread; trivial)
__global__ void offsets_kernel(const int* __restrict__ counts,
                               int* __restrict__ offsets,
                               int* __restrict__ cursor) {
    if (blockIdx.x == 0 && threadIdx.x == 0) {
        int a = 0;
        for (int e = 0; e < MOE_E; ++e) {
            offsets[e] = a;
            cursor[e] = a;
            a += counts[e];
        }
        offsets[MOE_E] = a;  // == 2*N
    }
}

// scatter token ids + gate weights into expert-grouped assignment lists
__global__ void scatter_kernel(const int* __restrict__ topE,
                               const float* __restrict__ topW,
                               int* __restrict__ cursor,
                               int* __restrict__ idx_list,
                               float* __restrict__ wgt_list) {
    int n = blockIdx.x * blockDim.x + threadIdx.x;
    if (n >= MOE_N) return;
    for (int k = 0; k < 2; ++k) {
        int e = topE[n * 2 + k];
        int slot = atomicAdd(&cursor[e], 1);
        idx_list[slot] = n;
        wgt_list[slot] = topW[n * 2 + k];
    }
}

// ---------------------------------------------------------------------------
// WMMA GEMM over gathered rows.
//   C[32 x 256 per block] = act(A[32 x K] @ Bt[e][Nout x K]^T + bias[e])
// grid = (Nout/256, 256 m-tiles, E); block = 256 (8 waves).
// Each wave: A frags {rows 0-15, 16-31} x B frags {cols +0, +16} -> 4 wmma.
// mode 0: relu -> bf16 store to Hout[slot][Nout]
// mode 1: scale by gate weight, atomicAdd f32 into out[token][Nout]
// ---------------------------------------------------------------------------
#define TILE_M 32
#define KCHUNK 512
#define APAD 8
#define ASTRIDE (KCHUNK + APAD)

__global__ __launch_bounds__(256) void moe_gemm_kernel(
    const __bf16* __restrict__ A,
    const __bf16* __restrict__ Bt,
    const float* __restrict__ bias,
    __bf16* __restrict__ Hout,
    float* __restrict__ out,
    const int* __restrict__ counts,
    const int* __restrict__ offsets,
    const int* __restrict__ idx_list,
    const float* __restrict__ wgt_list,
    int K, int Nout, int gather, int mode) {
    __shared__ __align__(16) __bf16 As[TILE_M * ASTRIDE];

    const int e = blockIdx.z;
    const int cnt = counts[e];
    const int m0 = blockIdx.y * TILE_M;
    if (m0 >= cnt) return;  // block-uniform: EXEC stays all-ones for WMMA
    const int base = offsets[e];

    const int lane = threadIdx.x & 31;
    const int wave = threadIdx.x >> 5;
    const int sel = (lane >> 4) & 1;   // half-wave select (K offset 0 / 8)
    const int nn = lane & 15;          // fragment row/col within 16
    const int ncol0 = blockIdx.x * 256 + wave * 32;

    // cooperative A-tile load mapping: thread -> (row, 64-elem span)
    const int lr = threadIdx.x >> 3;         // 0..31
    const int lc = (threadIdx.x & 7) * 64;   // 0..448
    const int srow = min(m0 + lr, cnt - 1);  // clamp (results masked later)
    const __bf16* arow = gather
        ? (A + (size_t)idx_list[base + srow] * K)
        : (A + (size_t)(base + srow) * K);

    v8f acc00 = {}, acc01 = {}, acc10 = {}, acc11 = {};

    for (int k0 = 0; k0 < K; k0 += KCHUNK) {
        // stage 32 x KCHUNK bf16 A tile into LDS (128B per thread)
        {
            const __bf16* srcp = arow + k0 + lc;
            __bf16* dstp = &As[lr * ASTRIDE + lc];
#if HAVE_ASYNC_LDS
#pragma unroll
            for (int i = 0; i < 8; ++i)
                __builtin_amdgcn_global_load_async_to_lds_b128(
                    (as1_v4i_p)(srcp + i * 8), (as3_v4i_p)(dstp + i * 8),
                    0, 0);
            async_wait0();
#else
            const uint4* s4 = (const uint4*)srcp;
            uint4* d4 = (uint4*)dstp;
#pragma unroll
            for (int i = 0; i < 8; ++i) d4[i] = s4[i];
#endif
        }
        __syncthreads();

#pragma unroll 2
        for (int ks = 0; ks < KCHUNK; ks += 32) {
            FragB16 a0, a1, b0, b1;
            // A fragments: lane nn holds row nn (a0) / 16+nn (a1)
            {
                const uint4* ap0 =
                    (const uint4*)(&As[nn * ASTRIDE + ks + 8 * sel]);
                a0.q[0] = ap0[0];
                a0.q[1] = ap0[2];  // +16 elements
                const uint4* ap1 =
                    (const uint4*)(&As[(16 + nn) * ASTRIDE + ks + 8 * sel]);
                a1.q[0] = ap1[0];
                a1.q[1] = ap1[2];
            }
            // B fragments: lane nn holds column (ncol0+nn [+16]), same K layout
            {
                size_t be = ((size_t)e * Nout + (ncol0 + nn)) * K +
                            (size_t)(k0 + ks + 8 * sel);
                const uint4* bp = (const uint4*)(Bt + be);
                b0.q[0] = bp[0];
                b0.q[1] = bp[2];
                const uint4* bp1 = (const uint4*)(Bt + be + (size_t)16 * K);
                b1.q[0] = bp1[0];
                b1.q[1] = bp1[2];
                __builtin_prefetch(Bt + be + 32, 0, 0);
            }
            acc00 = __builtin_amdgcn_wmma_f32_16x16x32_bf16(
                false, a0.v, false, b0.v, (short)0, acc00, false, false);
            acc01 = __builtin_amdgcn_wmma_f32_16x16x32_bf16(
                false, a0.v, false, b1.v, (short)0, acc01, false, false);
            acc10 = __builtin_amdgcn_wmma_f32_16x16x32_bf16(
                false, a1.v, false, b0.v, (short)0, acc10, false, false);
            acc11 = __builtin_amdgcn_wmma_f32_16x16x32_bf16(
                false, a1.v, false, b1.v, (short)0, acc11, false, false);
        }
        __syncthreads();
    }

    // epilogue: D layout -> lane holds (m = 16*afrag + r + 8*sel,
    //                                   n = ncol0 + nn + 16*bfrag)
    const int valid = cnt - m0;
    const float bias0 = bias[e * Nout + ncol0 + nn];
    const float bias1 = bias[e * Nout + ncol0 + 16 + nn];
#pragma unroll
    for (int r = 0; r < 8; ++r) {
        const int mrow = r + 8 * sel;
#pragma unroll
        for (int af = 0; af < 2; ++af) {
            int m = af * 16 + mrow;
            if (m >= valid) continue;
            int slot = base + m0 + m;
            float v0 = (af == 0 ? acc00[r] : acc10[r]) + bias0;
            float v1 = (af == 0 ? acc01[r] : acc11[r]) + bias1;
            if (mode == 0) {
                Hout[(size_t)slot * Nout + ncol0 + nn] =
                    (__bf16)fmaxf(v0, 0.0f);
                Hout[(size_t)slot * Nout + ncol0 + 16 + nn] =
                    (__bf16)fmaxf(v1, 0.0f);
            } else {
                int tok = idx_list[slot];
                float w = wgt_list[slot];
                atomicAdd(&out[(size_t)tok * Nout + ncol0 + nn],      w * v0);
                atomicAdd(&out[(size_t)tok * Nout + ncol0 + 16 + nn], w * v1);
            }
        }
    }
}

// ---------------------------------------------------------------------------
// Host launch
// ---------------------------------------------------------------------------
extern "C" void kernel_launch(void* const* d_in, const int* in_sizes, int n_in,
                              void* d_out, int out_size, void* d_ws,
                              size_t ws_size, hipStream_t stream) {
    const float* x   = (const float*)d_in[0];
    const float* W1  = (const float*)d_in[1];
    const float* b1  = (const float*)d_in[2];
    const float* W2  = (const float*)d_in[3];
    const float* b2  = (const float*)d_in[4];
    const float* W3  = (const float*)d_in[5];
    const float* b3  = (const float*)d_in[6];
    const float* Wg1 = (const float*)d_in[7];
    const float* bg1 = (const float*)d_in[8];
    const float* Wg2 = (const float*)d_in[9];
    const float* bg2 = (const float*)d_in[10];
    float* out = (float*)d_out;

    // --- workspace carve-out (deterministic) ---
    char* p = (char*)d_ws;
    auto carve = [&](size_t bytes) {
        char* r = p;
        p += (bytes + 255) & ~(size_t)255;
        return r;
    };
    __bf16* xb  = (__bf16*)carve((size_t)MOE_N * MOE_D * 2);
    __bf16* W1t = (__bf16*)carve((size_t)MOE_E * MOE_H * MOE_D * 2);
    __bf16* W2t = (__bf16*)carve((size_t)MOE_E * MOE_H * MOE_H * 2);
    __bf16* W3t = (__bf16*)carve((size_t)MOE_E * MOE_O * MOE_H * 2);
    __bf16* H1  = (__bf16*)carve((size_t)2 * MOE_N * MOE_H * 2);
    __bf16* H2  = (__bf16*)carve((size_t)2 * MOE_N * MOE_H * 2);
    int*   counts  = (int*)carve(32 * sizeof(int));  // counts[8],offsets[9],cursor[8]
    int*   offsets = counts + 8;
    int*   cursor  = counts + 20;
    int*   topE    = (int*)carve((size_t)2 * MOE_N * sizeof(int));
    float* topW    = (float*)carve((size_t)2 * MOE_N * sizeof(float));
    int*   idx_list = (int*)carve((size_t)2 * MOE_N * sizeof(int));
    float* wgt_list = (float*)carve((size_t)2 * MOE_N * sizeof(float));

    (void)hipMemsetAsync(out, 0, (size_t)MOE_N * MOE_O * sizeof(float), stream);
    (void)hipMemsetAsync(counts, 0, 8 * sizeof(int), stream);

    // precision conversion / weight transposition (one-shot, bandwidth-cheap)
    {
        size_t n = (size_t)MOE_N * MOE_D;
        convert_bf16_kernel<<<(unsigned)((n + 255) / 256), 256, 0, stream>>>(x, xb, n);
    }
    {
        size_t n = (size_t)MOE_E * MOE_D * MOE_H;
        transpose_bf16_kernel<<<(unsigned)((n + 255) / 256), 256, 0, stream>>>(
            W1, W1t, MOE_E, MOE_D, MOE_H);
    }
    {
        size_t n = (size_t)MOE_E * MOE_H * MOE_H;
        transpose_bf16_kernel<<<(unsigned)((n + 255) / 256), 256, 0, stream>>>(
            W2, W2t, MOE_E, MOE_H, MOE_H);
    }
    {
        size_t n = (size_t)MOE_E * MOE_H * MOE_O;
        transpose_bf16_kernel<<<(unsigned)((n + 255) / 256), 256, 0, stream>>>(
            W3, W3t, MOE_E, MOE_H, MOE_O);
    }

    // gating + routing
    gate_kernel<<<MOE_N, GATE_H, 0, stream>>>(x, Wg1, bg1, Wg2, bg2,
                                              counts, topE, topW);
    offsets_kernel<<<1, 1, 0, stream>>>(counts, offsets, cursor);
    scatter_kernel<<<MOE_N / 256, 256, 0, stream>>>(topE, topW, cursor,
                                                    idx_list, wgt_list);

    const int MAX_MTILES = MOE_N / TILE_M;  // 256 (worst case: all tokens one expert)

    // stage 1: H1 = relu(xb[gathered] @ W1[e] + b1[e])   K=D, Nout=H
    {
        dim3 grid(MOE_H / 256, MAX_MTILES, MOE_E);
        moe_gemm_kernel<<<grid, 256, 0, stream>>>(
            xb, W1t, b1, H1, nullptr, counts, offsets, idx_list, wgt_list,
            MOE_D, MOE_H, /*gather=*/1, /*mode=*/0);
    }
    // stage 2: H2 = relu(H1 @ W2[e] + b2[e])             K=H, Nout=H
    {
        dim3 grid(MOE_H / 256, MAX_MTILES, MOE_E);
        moe_gemm_kernel<<<grid, 256, 0, stream>>>(
            H1, W2t, b2, H2, nullptr, counts, offsets, idx_list, wgt_list,
            MOE_H, MOE_H, /*gather=*/0, /*mode=*/0);
    }
    // stage 3: out[tok] += w * (H2 @ W3[e] + b3[e])      K=H, Nout=O
    {
        dim3 grid(MOE_O / 256, MAX_MTILES, MOE_E);
        moe_gemm_kernel<<<grid, 256, 0, stream>>>(
            H2, W3t, b3, nullptr, out, counts, offsets, idx_list, wgt_list,
            MOE_H, MOE_O, /*gather=*/0, /*mode=*/1);
    }
}
